// LSTMLayer_7610682049213
// MI455X (gfx1250) — matrix operand order repaired and checked by reading, not compiled, not run on hardware
//
#include <hip/hip_runtime.h>
#include <stdint.h>

typedef __attribute__((ext_vector_type(16))) __bf16 bf16x16;
typedef __attribute__((ext_vector_type(8)))  float  f32x8;
typedef __attribute__((ext_vector_type(4)))  float  f32x4;
typedef __attribute__((ext_vector_type(4)))  unsigned int u32x4;
typedef __attribute__((ext_vector_type(8)))  int i32x8;
typedef __attribute__((ext_vector_type(4)))  int i32x4;
typedef __attribute__((address_space(3))) unsigned char lds_byte;

namespace {
constexpr int T_   = 512;
constexpr int B_   = 64;
constexpr int I_   = 256;
constexpr int H_   = 512;
constexpr int NBLK = 16;

// LDS layout (per block)
constexpr int WHH_STRIDE  = H_ * 2 + 16;                    // 1040 B per gate-column
constexpr int WIH_STRIDE  = I_ * 2 + 16;                    // 528 B per gate-column
constexpr int X_STRIDE    = I_ * 2 + 16;                    // 528 B per batch row
constexpr int X_BUF       = B_ * X_STRIDE;                  // 33792 B
constexpr int LDS_WIH_OFF  = 128 * WHH_STRIDE;              // 133120
constexpr int LDS_GATE_OFF = LDS_WIH_OFF + 128 * WIH_STRIDE;// 200704
constexpr int LDS_BIAS_OFF = LDS_GATE_OFF + 32768;          // 233472
constexpr int LDS_X_OFF    = LDS_BIAS_OFF + 512;            // 233984
constexpr int LDS_TOTAL    = LDS_X_OFF + 2 * X_BUF;         // 301568 < 320KB

// workspace layout (bytes)
constexpr size_t OFF_X    = 0;
constexpr size_t OFF_WHH  = OFF_X   + (size_t)T_ * B_ * I_ * 2;   // x bf16: 16 MB
constexpr size_t OFF_WIH  = OFF_WHH + (size_t)4 * H_ * H_ * 2;    // w_hh bf16: 2 MB
constexpr size_t OFF_HBUF = OFF_WIH + (size_t)4 * H_ * I_ * 2;    // w_ih bf16: 1 MB
constexpr size_t OFF_CNT  = OFF_HBUF + (size_t)2 * B_ * H_ * 2;   // h dbl-buf bf16
} // namespace

union Frag { u32x4 u[2]; bf16x16 v; };

__device__ __forceinline__ float sigmoidf_(float x) {
  return 1.0f / (1.0f + __expf(-x));
}
__device__ __forceinline__ float tanhf_(float x) {
  return 1.0f - 2.0f / (__expf(2.0f * x) + 1.0f);
}
__device__ __forceinline__ uint16_t f2bf(float f) {
  unsigned u = __float_as_uint(f);
  u += 0x7FFFu + ((u >> 16) & 1u);        // round-to-nearest-even
  return (uint16_t)(u >> 16);
}

// LDS byte offset of a __shared__ pointer (generic -> AS3 -> int)
__device__ __forceinline__ unsigned lds_off(const void* p) {
  return (unsigned)(unsigned long long)(lds_byte*)p;
}

// Issue one TDM 2D tile load (global -> LDS) with per-row LDS padding.
// tile: tile1 rows x tile0 elements (bf16), row stride stride0 elements.
// pad_interval: code (7 => every 1024B, 6 => every 512B); pad_amount: code 3 => 4 dwords (16B).
__device__ __forceinline__ void tdm_load_2d(unsigned lds_addr, const void* gaddr,
                                            unsigned td0, unsigned td1,
                                            unsigned tile0, unsigned tile1,
                                            unsigned stride0,
                                            unsigned pad_interval, unsigned pad_amount)
{
  unsigned long long ga = (unsigned long long)(uintptr_t)gaddr;
  u32x4 g0;
  g0[0] = 1u;                                            // count=1, user mode
  g0[1] = lds_addr;                                      // lds_addr[31:0]
  g0[2] = (unsigned)(ga & 0xFFFFFFFFu);                  // global_addr[31:0]
  g0[3] = (unsigned)((ga >> 32) & 0x01FFFFFFu) | 0x80000000u;  // ga[56:32] | type=2
  i32x8 g1;
  g1[0] = (int)((1u << 16) |                             // data_size = 2B
                (1u << 20) |                             // pad_enable
                (pad_interval << 22) | (pad_amount << 25));
  g1[1] = (int)((td0 & 0xFFFFu) << 16);                  // tensor_dim0[15:0]
  g1[2] = (int)(((td0 >> 16) & 0xFFFFu) | ((td1 & 0xFFFFu) << 16));
  g1[3] = (int)(((td1 >> 16) & 0xFFFFu) | ((tile0 & 0xFFFFu) << 16));
  g1[4] = (int)(tile1 & 0xFFFFu);                        // tile_dim1 (tile_dim2=0)
  g1[5] = (int)stride0;                                  // tensor_dim0_stride[31:0]
  g1[6] = 0;
  g1[7] = 0;                                             // dim1_stride unused (2D)
  i32x4 z4 = {0, 0, 0, 0};
#if defined(__clang_major__) && (__clang_major__ >= 23)
  i32x8 z8 = {0, 0, 0, 0, 0, 0, 0, 0};
  __builtin_amdgcn_tensor_load_to_lds(g0, g1, z4, z4, z8, 0);
#else
  __builtin_amdgcn_tensor_load_to_lds(g0, g1, z4, z4, 0);
#endif
}

__global__ void cvt_bf16(const float* __restrict__ src, uint16_t* __restrict__ dst, int n) {
  int i = blockIdx.x * blockDim.x + threadIdx.x;
  int stride = gridDim.x * blockDim.x;
  for (; i < n; i += stride) dst[i] = f2bf(src[i]);
}

__global__ void zero_cnt(unsigned int* c) {
  if (threadIdx.x == 0) *c = 0u;
}

// Persistent fused LSTM: 16 blocks x 256 threads (8 waves of 32).
// Block `blk` owns H columns [32*blk, 32*blk+32) for all 4 gates (128 gate cols).
// Wave w: gate g = w>>1, h-halftile hl = w&1; computes 4 M-tiles (batch 64 = 4x16).
// Per-step GEMM: [64 x 768] @ [768 x 128], K = H (recurrent, global) + I (x_t, TDM->LDS).
__global__ __launch_bounds__(256, 1) void lstm_persist(
    const uint16_t* __restrict__ xb,    // [T][B][I]  bf16
    const uint16_t* __restrict__ whhb,  // [4H][H]    bf16
    const uint16_t* __restrict__ wihb,  // [4H][I]    bf16
    uint16_t* hbuf,                     // [2][B][H]  bf16 double buffer
    const float* __restrict__ c0,
    const float* __restrict__ b_ih,
    const float* __restrict__ b_hh,
    float* __restrict__ out,            // [T][B][H] ++ h_f[B][H] ++ c_f[B][H]
    unsigned int* cnt)
{
  __shared__ __align__(16) unsigned char smem[LDS_TOTAL];
  const int tid = threadIdx.x;
  const int blk = blockIdx.x;
  const unsigned smem0 = lds_off(smem);

  // ---- one-time: weight slices -> LDS via Tensor Data Mover (wave 0 issues) ----
  if (tid < 32) {
#pragma unroll
    for (int gg = 0; gg < 4; ++gg) {
      const int grow = gg * H_ + 32 * blk;
      // W_hh gate slice: 32 rows x 512 bf16, +16B LDS pad per row (stride 1040)
      tdm_load_2d(smem0 + gg * 32 * WHH_STRIDE, whhb + (size_t)grow * H_,
                  H_, 32, H_, 32, H_, /*pad*/ 7, 3);
      // W_ih gate slice: 32 rows x 256 bf16, +16B LDS pad per row (stride 528)
      tdm_load_2d(smem0 + LDS_WIH_OFF + gg * 32 * WIH_STRIDE, wihb + (size_t)grow * I_,
                  I_, 32, I_, 32, I_, /*pad*/ 6, 3);
    }
    // x_0 tile: 64 rows x 256 bf16 -> padded rows (stride 528) in x buffer 0
    tdm_load_2d(smem0 + LDS_X_OFF, xb, I_, B_, I_, B_, I_, /*pad*/ 6, 3);
    __builtin_amdgcn_s_wait_tensorcnt(0);
  }
  // bias sums (parallel with TDM issue)
  if (tid < 128) {
    const int grow = (tid >> 5) * H_ + 32 * blk + (tid & 31);
    ((float*)(smem + LDS_BIAS_OFF))[tid] = b_ih[grow] + b_hh[grow];
  }
  __syncthreads();

  // ---- per-wave WMMA roles ----
  const int wv   = tid >> 5;
  const int g    = wv >> 1;      // gate 0..3 (i,f,g,o)
  const int hl   = wv & 1;       // h half-tile 0..1
  const int lane = tid & 31;
  const int n    = lane & 15;    // B col / A row within tile
  const int half = lane >> 4;
  const unsigned char* bHH = smem + (g * 32 + hl * 16 + n) * WHH_STRIDE;
  const unsigned char* bIH = smem + LDS_WIH_OFF + (g * 32 + hl * 16 + n) * WIH_STRIDE;
  float* gateL = (float*)(smem + LDS_GATE_OFF);
  const float bsum = ((const float*)(smem + LDS_BIAS_OFF))[g * 32 + hl * 16 + n];

  // ---- per-thread elementwise role: 8 fixed cells, c stays in registers ----
  const int em   = tid >> 6;           // m tile 0..3
  const int eht  = (tid >> 5) & 1;     // h tile 0..1
  const int erow = (tid >> 1) & 15;
  const int ecol = (tid & 1) * 8;
  const int ebb  = em * 16 + erow;                 // batch index
  const int ehg  = 32 * blk + eht * 16 + ecol;     // global h col (8 consecutive)

  float creg[8];
  {
    const f32x4* cp = (const f32x4*)(c0 + (size_t)ebb * H_ + ehg);
    f32x4 a0 = cp[0], a1 = cp[1];
    creg[0] = a0.x; creg[1] = a0.y; creg[2] = a0.z; creg[3] = a0.w;
    creg[4] = a1.x; creg[5] = a1.y; creg[6] = a1.z; creg[7] = a1.w;
  }

  for (int t = 0; t < T_; ++t) {
    const uint16_t* hin = hbuf + (size_t)(t & 1) * (B_ * H_);
    uint16_t* hob = hbuf + (size_t)((t + 1) & 1) * (B_ * H_);
    const unsigned char* xl = smem + LDS_X_OFF + (t & 1) * X_BUF;

    // kick off async TDM prefetch of x_{t+1}; waited at the step barrier
    if (tid < 32 && (t + 1) < T_) {
      tdm_load_2d(smem0 + LDS_X_OFF + ((t + 1) & 1) * X_BUF,
                  xb + (size_t)(t + 1) * (B_ * I_), I_, B_, I_, B_, I_, 6, 3);
    }

    f32x8 acc[4] = {};
    // recurrent K: h(t-1) @ W_hh^T slice, K = 512 (16 x k32), A from global/L2
#pragma unroll 4
    for (int kt = 0; kt < 16; ++kt) {
      Frag bf_;
      const u32x4* bp = (const u32x4*)(bHH + (kt * 32 + half * 8) * 2);
      bf_.u[0] = bp[0]; bf_.u[1] = bp[2];
#pragma unroll
      for (int m = 0; m < 4; ++m) {
        Frag af;
        const uint16_t* ap = hin + (size_t)(m * 16 + n) * H_ + kt * 32 + half * 8;
        af.u[0] = *(const u32x4*)ap;
        af.u[1] = *(const u32x4*)(ap + 16);
        acc[m] = __builtin_amdgcn_wmma_f32_16x16x32_bf16(
            false, af.v, false, bf_.v, (short)0, acc[m], false, false);
      }
    }
    // input K: x_t @ W_ih^T slice, K = 256 (8 x k32), A from LDS (TDM-prefetched)
#pragma unroll 4
    for (int kt = 0; kt < 8; ++kt) {
      Frag bf_;
      const u32x4* bp = (const u32x4*)(bIH + (kt * 32 + half * 8) * 2);
      bf_.u[0] = bp[0]; bf_.u[1] = bp[2];
#pragma unroll
      for (int m = 0; m < 4; ++m) {
        Frag af;
        const unsigned char* ap = xl + (size_t)(m * 16 + n) * X_STRIDE + (kt * 32 + half * 8) * 2;
        af.u[0] = *(const u32x4*)ap;
        af.u[1] = *(const u32x4*)(ap + 32);
        acc[m] = __builtin_amdgcn_wmma_f32_16x16x32_bf16(
            false, af.v, false, bf_.v, (short)0, acc[m], false, false);
      }
    }

    // stage gate tiles (+bias) to LDS, gate-interleaved for float4 readback
#pragma unroll
    for (int m = 0; m < 4; ++m) {
#pragma unroll
      for (int r = 0; r < 8; ++r) {
        gateL[(((m * 2 + hl) * 256) + (r + 8 * half) * 16 + n) * 4 + g] = acc[m][r] + bsum;
      }
    }
    __syncthreads();

    // cell update: c' = sig(f)*c + sig(i)*tanh(g); h' = sig(o)*tanh(c')
    const float* g4 = gateL + (((em * 2 + eht) * 256) + erow * 16 + ecol) * 4;
    float hn8[8];
#pragma unroll
    for (int j = 0; j < 8; ++j) {
      float iv = g4[j * 4 + 0];
      float fv = g4[j * 4 + 1];
      float gv = g4[j * 4 + 2];
      float ov = g4[j * 4 + 3];
      float cn = sigmoidf_(fv) * creg[j] + sigmoidf_(iv) * tanhf_(gv);
      creg[j] = cn;
      hn8[j]  = sigmoidf_(ov) * tanhf_(cn);
    }

    float* orow = out + (size_t)t * (B_ * H_) + (size_t)ebb * H_ + ehg;
#pragma unroll
    for (int j = 0; j < 8; ++j) orow[j] = hn8[j];

    {   // bf16 h for next step's A matrix
      union { uint16_t s[8]; u32x4 u; } hb;
#pragma unroll
      for (int j = 0; j < 8; ++j) hb.s[j] = f2bf(hn8[j]);
      *(u32x4*)(hob + (size_t)ebb * H_ + ehg) = hb.u;
    }

    if (t == T_ - 1) {
      float* hf = out + (size_t)T_ * B_ * H_ + (size_t)ebb * H_ + ehg;
      float* cf = hf + (size_t)B_ * H_;
#pragma unroll
      for (int j = 0; j < 8; ++j) { hf[j] = hn8[j]; cf[j] = creg[j]; }
    }

    // ---- device-wide step barrier (release h stores, acquire for next reads) ----
    __threadfence();
    __syncthreads();
    if (tid == 0) {
      __hip_atomic_fetch_add(cnt, 1u, __ATOMIC_ACQ_REL, __HIP_MEMORY_SCOPE_AGENT);
      const unsigned int target = (unsigned)NBLK * (unsigned)(t + 1);
      while (__hip_atomic_load(cnt, __ATOMIC_ACQUIRE, __HIP_MEMORY_SCOPE_AGENT) < target)
        __builtin_amdgcn_s_sleep(2);
    }
    if (tid < 32) __builtin_amdgcn_s_wait_tensorcnt(0);   // x_{t+1} DMA landed
    __syncthreads();
  }
}

extern "C" void kernel_launch(void* const* d_in, const int* in_sizes, int n_in,
                              void* d_out, int out_size, void* d_ws, size_t ws_size,
                              hipStream_t stream) {
  (void)in_sizes; (void)n_in; (void)out_size; (void)ws_size;
  const float* x    = (const float*)d_in[0];
  const float* h0   = (const float*)d_in[1];
  const float* c0   = (const float*)d_in[2];
  const float* w_ih = (const float*)d_in[3];
  const float* w_hh = (const float*)d_in[4];
  const float* b_ih = (const float*)d_in[5];
  const float* b_hh = (const float*)d_in[6];

  uint8_t* ws = (uint8_t*)d_ws;
  uint16_t* xb   = (uint16_t*)(ws + OFF_X);
  uint16_t* whhb = (uint16_t*)(ws + OFF_WHH);
  uint16_t* wihb = (uint16_t*)(ws + OFF_WIH);
  uint16_t* hbuf = (uint16_t*)(ws + OFF_HBUF);
  unsigned int* cnt = (unsigned int*)(ws + OFF_CNT);
  float* out = (float*)d_out;

  cvt_bf16<<<1024, 256, 0, stream>>>(x,    xb,   T_ * B_ * I_);
  cvt_bf16<<<256,  256, 0, stream>>>(w_hh, whhb, 4 * H_ * H_);
  cvt_bf16<<<256,  256, 0, stream>>>(w_ih, wihb, 4 * H_ * I_);
  cvt_bf16<<<64,   256, 0, stream>>>(h0,   hbuf, B_ * H_);   // parity-0 buffer
  zero_cnt<<<1, 32, 0, stream>>>(cnt);
  lstm_persist<<<NBLK, 256, 0, stream>>>(xb, whhb, wihb, hbuf, c0, b_ih, b_hh, out, cnt);
}
